// NewsClassifier_23527830847606
// MI455X (gfx1250) — matrix operand order repaired and checked by reading, not compiled, not run on hardware
//
#include <hip/hip_runtime.h>
#include <hip/hip_bf16.h>

typedef __bf16 bf16_t;
typedef __attribute__((ext_vector_type(16))) __bf16 bf16x16;
typedef __attribute__((ext_vector_type(8)))  float  f32x8;

#define HD 256   // hidden dim

// ---------------------------------------------------------------------------
// WMMA GEMM: C[M,256] = A[M,K](f32, row-major) * B[K,256]
// B supplied pre-transposed as Bt[256,K] bf16 (row n of Bt = column n of B).
// A converted f32->bf16 in-register. v_wmma_f32_16x16x32_bf16, f32 accum.
// Block = 128 threads = 4 waves; each wave computes a 16x128 strip of C
// (8 n-tiles, 8x v8f accumulators); block covers 32 rows x 256 cols.
// A is read only twice per GEMM (once per 128-col half).
// M must be a multiple of 32; K a multiple of 32. No divergence: EXEC
// all-ones for WMMA.
// ---------------------------------------------------------------------------
__global__ __launch_bounds__(128)
void wmma_gemm_bf16(const float* __restrict__ A, const bf16_t* __restrict__ Bt,
                    float* __restrict__ C, int M, int K) {
  const int lane  = threadIdx.x & 31;
  const int wave  = threadIdx.x >> 5;
  const int half  = lane >> 4;     // 0 or 1
  const int l16   = lane & 15;
  const int mtile = blockIdx.x * 2 + (wave >> 1);   // 16-row tile index
  const int nt0   = (wave & 1) * 8;                 // first of 8 n-tiles
  const float* arow = A + (size_t)(mtile * 16 + l16) * K;

  f32x8 acc[8] = {f32x8{}, f32x8{}, f32x8{}, f32x8{},
                  f32x8{}, f32x8{}, f32x8{}, f32x8{}};

  for (int k0 = 0; k0 < K; k0 += 32) {
    // A fragment: 16-bit A 16x32 layout. lanes 0-15: rows 0-15, K groups
    // {k0+half*8 .. +7} then {k0+16+half*8 .. +7} (pairs packed per VGPR).
    bf16x16 a;
    const float* p0 = arow + k0 + half * 8;
    const float* p1 = p0 + 16;
#pragma unroll
    for (int i = 0; i < 8; ++i) a[i]     = (bf16_t)p0[i];
#pragma unroll
    for (int i = 0; i < 8; ++i) a[8 + i] = (bf16_t)p1[i];

#pragma unroll
    for (int t = 0; t < 8; ++t) {
      // B fragment: lane holds column (nt0+t)*16+l16; K = k0 + half*8 + e (e<8), k0 + 16 + half*8 + (e-8) (e>=8).
      const bf16_t* bp = Bt + (size_t)((nt0 + t) * 16 + l16) * K + k0 + half * 8;
      bf16x16 b;
#pragma unroll
      for (int i = 0; i < 16; ++i) b[i] = bp[i < 8 ? i : i + 8];
      acc[t] = __builtin_amdgcn_wmma_f32_16x16x32_bf16(
          false, a, false, b, (short)0, acc[t], false, false);
    }
  }

  // D layout: VGPR r -> row (half*8 + r), col = l16 within tile.
#pragma unroll
  for (int t = 0; t < 8; ++t) {
    const int col = (nt0 + t) * 16 + l16;
#pragma unroll
    for (int r = 0; r < 8; ++r) {
      const int m = mtile * 16 + half * 8 + r;
      C[(size_t)m * HD + col] = acc[t][r];
    }
  }
}

// ---------------------------------------------------------------------------
// Helpers
// ---------------------------------------------------------------------------
__global__ void k_fill_f32(float* p, float v, int n) {
  int i = blockIdx.x * blockDim.x + threadIdx.x;
  if (i < n) p[i] = v;
}

// Wt[n*K + k] = (bf16) W[k*N + n]  (transpose + convert)
__global__ void k_cvt_transpose(const float* __restrict__ W, bf16_t* __restrict__ Wt,
                                int K, int N) {
  int i = blockIdx.x * blockDim.x + threadIdx.x;
  if (i >= K * N) return;
  int n = i / K, k = i % K;
  Wt[i] = (bf16_t)W[(size_t)k * N + n];
}

__global__ void k_deg_accum(const int* __restrict__ dst, float* deg, int E) {
  int i = blockIdx.x * blockDim.x + threadIdx.x;
  if (i < E) unsafeAtomicAdd(&deg[dst[i]], 1.0f);
}

__global__ void k_rsqrt_inplace(float* p, int n) {
  int i = blockIdx.x * blockDim.x + threadIdx.x;
  if (i < n) p[i] = rsqrtf(p[i]);
}

// out[n,h] = h_in[n,h] * dinv[n]^2   (self-loop term initializes accumulator)
__global__ void k_selfloop_init(const float* __restrict__ h_in,
                                const float* __restrict__ dinv,
                                float* __restrict__ out, int total) {
  int i = blockIdx.x * blockDim.x + threadIdx.x;
  if (i >= total) return;
  float di = dinv[i >> 8];
  out[i] = h_in[i] * di * di;
}

// one wave32 per edge: out[dst] += h[src] * dinv[src]*dinv[dst]  (256 floats)
__global__ void k_edge_agg(const float* __restrict__ h, const int* __restrict__ src,
                           const int* __restrict__ dst, const float* __restrict__ dinv,
                           float* __restrict__ out, int E) {
  int wid  = (int)((blockIdx.x * (size_t)blockDim.x + threadIdx.x) >> 5);
  int lane = threadIdx.x & 31;
  if (wid >= E) return;
  int s = src[wid], d = dst[wid];
  float coef = dinv[s] * dinv[d];
  const float* hs = h + (size_t)s * HD;
  float* od = out + (size_t)d * HD;
#pragma unroll
  for (int j = 0; j < 8; ++j) {
    int idx = lane + j * 32;
    unsafeAtomicAdd(od + idx, hs[idx] * coef);
  }
}

__global__ void k_relu_inplace(float* p, int total) {
  int i = blockIdx.x * blockDim.x + threadIdx.x;
  if (i < total) p[i] = fmaxf(p[i], 0.0f);
}

__global__ void k_pool_accum(const float* __restrict__ x, const int* __restrict__ seg,
                             float* __restrict__ sum, int total) {
  int i = blockIdx.x * blockDim.x + threadIdx.x;
  if (i >= total) return;
  unsafeAtomicAdd(&sum[(size_t)seg[i >> 8] * HD + (i & 255)], x[i]);
}

__global__ void k_count_accum(const int* __restrict__ seg, float* cnt, int n) {
  int i = blockIdx.x * blockDim.x + threadIdx.x;
  if (i < n) unsafeAtomicAdd(&cnt[seg[i]], 1.0f);
}

__global__ void k_pool_div(const float* __restrict__ sum, const float* __restrict__ cnt,
                           float* __restrict__ out, int total) {
  int i = blockIdx.x * blockDim.x + threadIdx.x;
  if (i >= total) return;
  out[i] = sum[i] / fmaxf(cnt[i >> 8], 1.0f);
}

// reps[64,1280] = concat(content, repost, comment, video, image) reps [64,256] each
__global__ void k_concat(const float* c, const float* r, const float* cm,
                         const float* v, const float* im, float* reps) {
  int i = blockIdx.x * blockDim.x + threadIdx.x;
  if (i >= 64 * 1280) return;
  int row = i / 1280, col = i % 1280;
  int sel = col >> 8, cc = col & 255;
  const float* s = (sel == 0) ? c : (sel == 1) ? r : (sel == 2) ? cm
                 : (sel == 3) ? v : im;
  reps[i] = s[row * HD + cc];
}

__global__ void k_mlp1(const float* __restrict__ reps, const float* __restrict__ W1,
                       const float* __restrict__ b1, float* __restrict__ h) {
  int i = blockIdx.x * blockDim.x + threadIdx.x;   // 64*256
  if (i >= 64 * HD) return;
  int row = i >> 8, j = i & 255;
  float acc = b1[j];
  const float* rp = reps + row * 1280;
  for (int k = 0; k < 1280; ++k) acc += rp[k] * W1[(size_t)k * HD + j];
  h[i] = fmaxf(acc, 0.0f);
}

__global__ void k_mlp2(const float* __restrict__ h, const float* __restrict__ W2,
                       const float* __restrict__ b2, float* __restrict__ preds) {
  int i = blockIdx.x * blockDim.x + threadIdx.x;   // 64*2
  if (i >= 64 * 2) return;
  int row = i >> 1, j = i & 1;
  float acc = b2[j];
  const float* hp = h + row * HD;
  for (int k = 0; k < HD; ++k) acc += hp[k] * W2[k * 2 + j];
  preds[i] = acc;
}

__global__ void k_loss(const float* __restrict__ preds, const int* __restrict__ label,
                       float* __restrict__ out) {
  __shared__ float s[64];
  int i = threadIdx.x;  // 64 threads
  float p0 = preds[i * 2], p1 = preds[i * 2 + 1];
  float m = fmaxf(p0, p1);
  float lse = m + logf(__expf(p0 - m) + __expf(p1 - m));
  s[i] = (label[i] ? p1 : p0) - lse;
  __syncthreads();
  for (int st = 32; st > 0; st >>= 1) {
    if (i < st) s[i] += s[i + st];
    __syncthreads();
  }
  if (i == 0) out[0] = -s[0] / 64.0f;
}

// ---------------------------------------------------------------------------
extern "C" void kernel_launch(void* const* d_in, const int* in_sizes, int n_in,
                              void* d_out, int out_size, void* d_ws, size_t ws_size,
                              hipStream_t stream) {
  const int Bn = 64, D = 768, NR = 100000, ER = 400000, NC = 100000, EC = 300000, G = 512;

  const float* content   = (const float*)d_in[0];
  const float* video     = (const float*)d_in[1];
  const float* image     = (const float*)d_in[2];
  const float* repost_x  = (const float*)d_in[3];
  const float* comment_x = (const float*)d_in[4];
  const float* W_text  = (const float*)d_in[5];
  const float* W_video = (const float*)d_in[6];
  const float* W_image = (const float*)d_in[7];
  const float* Wr1 = (const float*)d_in[8];
  const float* Wr2 = (const float*)d_in[9];
  const float* Wc1 = (const float*)d_in[10];
  const float* Wc2 = (const float*)d_in[11];
  const float* W1 = (const float*)d_in[12];
  const float* b1 = (const float*)d_in[13];
  const float* W2 = (const float*)d_in[14];
  const float* b2 = (const float*)d_in[15];
  const int* repost_ei  = (const int*)d_in[16];
  const int* repost_batch = (const int*)d_in[17];
  const int* comment_ei = (const int*)d_in[18];
  const int* comment_batch = (const int*)d_in[19];
  const int* comment_graph_batch = (const int*)d_in[20];
  const int* label = (const int*)d_in[21];

  // workspace bump allocator (256B aligned)
  char* ws = (char*)d_ws;
  size_t off = 0;
  auto alloc = [&](size_t bytes) -> void* {
    void* p = ws + off;
    off = (off + bytes + 255) & ~(size_t)255;
    return p;
  };

  float*  buf1   = (float*)alloc((size_t)NR * HD * 4);
  float*  buf2   = (float*)alloc((size_t)NR * HD * 4);
  float*  dinv   = (float*)alloc((size_t)NR * 4);
  bf16_t* wt_txt = (bf16_t*)alloc((size_t)D * HD * 2);
  bf16_t* wt_vid = (bf16_t*)alloc((size_t)D * HD * 2);
  bf16_t* wt_img = (bf16_t*)alloc((size_t)D * HD * 2);
  bf16_t* wt_r1  = (bf16_t*)alloc((size_t)HD * HD * 2);
  bf16_t* wt_r2  = (bf16_t*)alloc((size_t)HD * HD * 2);
  bf16_t* wt_c1  = (bf16_t*)alloc((size_t)HD * HD * 2);
  bf16_t* wt_c2  = (bf16_t*)alloc((size_t)HD * HD * 2);
  float*  sumB   = (float*)alloc((size_t)Bn * HD * 4);
  float*  cntB   = (float*)alloc((size_t)Bn * 4);
  float*  sumG   = (float*)alloc((size_t)G * HD * 4);
  float*  cntG   = (float*)alloc((size_t)G * 4);
  float*  graphs = (float*)alloc((size_t)G * HD * 4);
  float*  rep_c  = (float*)alloc((size_t)Bn * HD * 4);   // content
  float*  rep_r  = (float*)alloc((size_t)Bn * HD * 4);   // repost
  float*  rep_m  = (float*)alloc((size_t)Bn * HD * 4);   // comment
  float*  rep_v  = (float*)alloc((size_t)Bn * HD * 4);   // video
  float*  rep_i  = (float*)alloc((size_t)Bn * HD * 4);   // image
  float*  reps   = (float*)alloc((size_t)Bn * 1280 * 4);
  float*  mlp_h  = (float*)alloc((size_t)Bn * HD * 4);
  (void)ws_size; (void)in_sizes; (void)n_in; (void)out_size;

  // --- weight transpose+convert (tiny) ---
  {
    int nDT = D * HD, nHH = HD * HD;
    k_cvt_transpose<<<(nDT + 255) / 256, 256, 0, stream>>>(W_text,  wt_txt, D,  HD);
    k_cvt_transpose<<<(nDT + 255) / 256, 256, 0, stream>>>(W_video, wt_vid, D,  HD);
    k_cvt_transpose<<<(nDT + 255) / 256, 256, 0, stream>>>(W_image, wt_img, D,  HD);
    k_cvt_transpose<<<(nHH + 255) / 256, 256, 0, stream>>>(Wr1, wt_r1, HD, HD);
    k_cvt_transpose<<<(nHH + 255) / 256, 256, 0, stream>>>(Wr2, wt_r2, HD, HD);
    k_cvt_transpose<<<(nHH + 255) / 256, 256, 0, stream>>>(Wc1, wt_c1, HD, HD);
    k_cvt_transpose<<<(nHH + 255) / 256, 256, 0, stream>>>(Wc2, wt_c2, HD, HD);
  }

  // --- modality projections via WMMA (M=64) ---
  wmma_gemm_bf16<<<Bn / 32, 128, 0, stream>>>(content, wt_txt, rep_c, Bn, D);
  wmma_gemm_bf16<<<Bn / 32, 128, 0, stream>>>(video,   wt_vid, rep_v, Bn, D);
  wmma_gemm_bf16<<<Bn / 32, 128, 0, stream>>>(image,   wt_img, rep_i, Bn, D);

  // --- GCN branch runner ---
  auto run_gcn = [&](const float* X, const bf16_t* Wt1, const bf16_t* Wt2,
                     const int* src, const int* dstv, int Nn, int E) {
    int totNH = Nn * HD;
    // x @ W_text  -> buf1
    wmma_gemm_bf16<<<Nn / 32, 128, 0, stream>>>(X, wt_txt, buf1, Nn, D);
    // degrees -> dinv
    k_fill_f32<<<(Nn + 255) / 256, 256, 0, stream>>>(dinv, 1.0f, Nn);
    k_deg_accum<<<(E + 255) / 256, 256, 0, stream>>>(dstv, dinv, E);
    k_rsqrt_inplace<<<(Nn + 255) / 256, 256, 0, stream>>>(dinv, Nn);
    // conv1 linear: buf2 = buf1 @ Wt1
    wmma_gemm_bf16<<<Nn / 32, 128, 0, stream>>>(buf1, Wt1, buf2, Nn, HD);
    // propagate into buf1 (self-loop init + edge scatter), relu
    k_selfloop_init<<<Nn, 256, 0, stream>>>(buf2, dinv, buf1, totNH);
    k_edge_agg<<<(E + 7) / 8, 256, 0, stream>>>(buf2, src, dstv, dinv, buf1, E);
    k_relu_inplace<<<Nn, 256, 0, stream>>>(buf1, totNH);
    // conv2
    wmma_gemm_bf16<<<Nn / 32, 128, 0, stream>>>(buf1, Wt2, buf2, Nn, HD);
    k_selfloop_init<<<Nn, 256, 0, stream>>>(buf2, dinv, buf1, totNH);
    k_edge_agg<<<(E + 7) / 8, 256, 0, stream>>>(buf2, src, dstv, dinv, buf1, E);
    // result in buf1 [Nn, HD]
  };

  // --- repost branch ---
  run_gcn(repost_x, wt_r1, wt_r2, repost_ei, repost_ei + ER, NR, ER);
  k_fill_f32<<<(Bn * HD + 255) / 256, 256, 0, stream>>>(sumB, 0.0f, Bn * HD);
  k_fill_f32<<<1, 64, 0, stream>>>(cntB, 0.0f, Bn);
  k_pool_accum<<<NR, 256, 0, stream>>>(buf1, repost_batch, sumB, NR * HD);
  k_count_accum<<<(NR + 255) / 256, 256, 0, stream>>>(repost_batch, cntB, NR);
  k_pool_div<<<(Bn * HD + 255) / 256, 256, 0, stream>>>(sumB, cntB, rep_r, Bn * HD);

  // --- comment branch ---
  run_gcn(comment_x, wt_c1, wt_c2, comment_ei, comment_ei + EC, NC, EC);
  k_fill_f32<<<(G * HD + 255) / 256, 256, 0, stream>>>(sumG, 0.0f, G * HD);
  k_fill_f32<<<(G + 255) / 256, 256, 0, stream>>>(cntG, 0.0f, G);
  k_pool_accum<<<NC, 256, 0, stream>>>(buf1, comment_batch, sumG, NC * HD);
  k_count_accum<<<(NC + 255) / 256, 256, 0, stream>>>(comment_batch, cntG, NC);
  k_pool_div<<<(G * HD + 255) / 256, 256, 0, stream>>>(sumG, cntG, graphs, G * HD);
  // graphs -> news items
  k_fill_f32<<<(Bn * HD + 255) / 256, 256, 0, stream>>>(sumB, 0.0f, Bn * HD);
  k_fill_f32<<<1, 64, 0, stream>>>(cntB, 0.0f, Bn);
  k_pool_accum<<<G, 256, 0, stream>>>(graphs, comment_graph_batch, sumB, G * HD);
  k_count_accum<<<(G + 255) / 256, 256, 0, stream>>>(comment_graph_batch, cntB, G);
  k_pool_div<<<(Bn * HD + 255) / 256, 256, 0, stream>>>(sumB, cntB, rep_m, Bn * HD);

  // --- head ---
  float* preds = (float*)d_out;          // [64,2]
  float* loss  = preds + Bn * 2;         // scalar
  k_concat<<<(Bn * 1280 + 255) / 256, 256, 0, stream>>>(rep_c, rep_r, rep_m, rep_v, rep_i, reps);
  k_mlp1<<<(Bn * HD + 255) / 256, 256, 0, stream>>>(reps, W1, b1, mlp_h);
  k_mlp2<<<1, 128, 0, stream>>>(mlp_h, W2, b2, preds);
  k_loss<<<1, 64, 0, stream>>>(preds, label, loss);
}
